// LowRankConv3D_36885179138595
// MI455X (gfx1250) — matrix operand order repaired and checked by reading, not compiled, not run on hardware
//
#include <hip/hip_runtime.h>
#include <hip/hip_bf16.h>
#include <stddef.h>

// ---------------------------------------------------------------------------
// CPD low-rank 3D conv, MI455X (gfx1250, wave32, WMMA).
//
// x[1,256,64^3] f32, rank R=256, Cout=256:
//   GEMM1 (c->r) -> separable 3-tap conv (h,w,d) -> GEMM2 (r->co) + bias.
// ~70 GFLOP vs >=512MB mandatory HBM traffic @ 23.3 TB/s => memory-bound.
//  - bf16 WMMA (f32 accum) for both channel GEMMs
//  - intermediate t1 stored bf16 (128MB) => L2-resident (192MB L2); the 9x
//    h/w-conv re-reads hit L2, not HBM
//  - d-axis conv fused into kernel 1 (d = contiguous axis)
//  - x tile staged into LDS via global_load_async_to_lds_b128 (ASYNCcnt)
//  - all hot-path global traffic is 16B-vectorized
// ---------------------------------------------------------------------------

typedef __attribute__((ext_vector_type(16))) __bf16 v16bf;
typedef __attribute__((ext_vector_type(8)))  __bf16 v8bf;
typedef __attribute__((ext_vector_type(8)))  float  v8f;
typedef __attribute__((ext_vector_type(4)))  float  v4f;

#define S      64
#define SP     (S * S * S)      // 262144 spatial positions
#define HW     (S * S)          // 4096 (h,w) lines
#define CIN    256
#define RNK    256
#define COUT   256
#define BSTR   272              // padded LDS bf16 row stride: 544B, 32B-aligned slices
#define ZSTR   68               // padded LDS f32 row stride

static __device__ __forceinline__ __bf16 f2bf(float f) { return (__bf16)f; }

// ---------------------------------------------------------------------------
// A-fragment swizzle (ISA 7.12.2, 16-bit A 16x32):
//   lane: half = lane>>4, m = lane&15
//   elem i: k = half*16 + i (contiguous, not the ISA table order; B matches)
// ---------------------------------------------------------------------------
__global__ __launch_bounds__(256) void lrconv_prep(const float* __restrict__ Ucin,
                                                   const float* __restrict__ Ucout,
                                                   __bf16* __restrict__ A1,
                                                   __bf16* __restrict__ A2) {
    int gid = blockIdx.x * blockDim.x + threadIdx.x;     // 0..8191
    if (gid >= 2 * 16 * 8 * 32) return;
    int which = gid >> 12;
    int rem   = gid & 4095;
    int t    = rem >> 8;
    int kc   = (rem >> 5) & 7;
    int lane = rem & 31;
    int half = lane >> 4, mrow = lane & 15;
    int m = t * 16 + mrow;
    const float* src = which ? Ucout : Ucin;              // both [K=256][M=256] row-major
    __bf16* dst = (which ? A2 : A1) + (size_t)rem * 16;
#pragma unroll
    for (int i = 0; i < 16; ++i) {
        int kloc = half * 16 + i;
        int k = kc * 32 + kloc;
        dst[i] = f2bf(src[k * 256 + m]);                  // A[m][k] = U[k][m]
    }
}

// ---------------------------------------------------------------------------
// Kernel 1: one workgroup per (h,w) line.
//   1) async-stage x[c, d=0..63] (f32, 64KB) into LDS  (ASYNCcnt path)
//   2) convert to bf16 B-fragment layout [d][c]
//   3) y[r,d] = sum_c Ucin[c,r] x[c,d]  via v_wmma_f32_16x16x32_bf16
//   4) d-axis 3-tap conv through LDS; 16B-vector bf16 store of t1 [hw][r][d]
// LDS: smem (69.6KB) time-multiplexed: x-f32 staging -> GEMM result Z.
// ---------------------------------------------------------------------------
__global__ __launch_bounds__(256) void lrconv_k1(const float* __restrict__ x,
                                                 const float* __restrict__ Ukd,
                                                 const __bf16* __restrict__ Afrag,
                                                 __bf16* __restrict__ t1) {
    __shared__ __align__(32) unsigned char smem[RNK * ZSTR * 4]; // 69632B
    __shared__ __align__(32) __bf16 Bt[S * BSTR];                // 34.8KB
    float* Xs = (float*)smem;                                    // [c][d] 64KB
    float* Z  = (float*)smem;                                    // [r][ZSTR]

    const int hw   = blockIdx.x;
    const int base = hw * S;
    const int tid  = threadIdx.x;
    const int lane = tid & 31;
    const int wave = tid >> 5;
    const int half = lane >> 4;
    const int ncol = lane & 15;

    // ---- 1) async copy x tile (64KB, 4096 x b128) into LDS ----
#pragma unroll
    for (int it = 0; it < 16; ++it) {
        int e  = tid + it * 256;
        int c  = e >> 4;
        int d4 = (e & 15) * 4;
        unsigned long long ga = (unsigned long long)(const void*)(x + (size_t)c * SP + base + d4);
        unsigned ldsoff = (unsigned)(size_t)(&Xs[(c << 6) + d4]);
        asm volatile("global_load_async_to_lds_b128 %0, %1, off"
                     :: "v"(ldsoff), "v"(ga) : "memory");
    }
    asm volatile("s_wait_asynccnt 0" ::: "memory");
    __syncthreads();

    // ---- 2) convert f32 -> bf16 B-fragment layout Bt[d][c] ----
    {
        int d = tid & 63;
        for (int c = tid >> 6; c < CIN; c += 4)
            Bt[d * BSTR + c] = f2bf(Xs[(c << 6) + d]);
    }
    __syncthreads();

    // ---- 3) GEMM1: 8 waves x 2 r-tiles, 4 d-column tiles, K=256 ----
    {
        v8f acc[2][4];
#pragma unroll
        for (int tt = 0; tt < 2; ++tt)
#pragma unroll
            for (int ct = 0; ct < 4; ++ct)
                acc[tt][ct] = v8f{0.f, 0.f, 0.f, 0.f, 0.f, 0.f, 0.f, 0.f};

#pragma unroll
        for (int kc = 0; kc < 8; ++kc) {
            v16bf b[4];                                   // shared across both r-tiles
#pragma unroll
            for (int ct = 0; ct < 4; ++ct)
                b[ct] = *(const v16bf*)(&Bt[(ct * 16 + ncol) * BSTR + kc * 32 + half * 16]);
#pragma unroll
            for (int tt = 0; tt < 2; ++tt) {
                const int t = wave * 2 + tt;
                const v16bf a = *(const v16bf*)(Afrag + ((size_t)(t * 8 + kc) * 32 + lane) * 16);
#pragma unroll
                for (int ct = 0; ct < 4; ++ct)
                    acc[tt][ct] = __builtin_amdgcn_wmma_f32_16x16x32_bf16(
                        false, a, false, b[ct], (short)0, acc[tt][ct], false, false);
            }
        }
        __syncthreads();                                  // Xs no longer needed; reuse as Z
#pragma unroll
        for (int tt = 0; tt < 2; ++tt) {
            const int t = wave * 2 + tt;
#pragma unroll
            for (int ct = 0; ct < 4; ++ct)
#pragma unroll
                for (int j = 0; j < 8; ++j) {
                    int r = t * 16 + j + 8 * half;        // C/D: VGPR j -> M = j + 8*half
                    Z[r * ZSTR + ct * 16 + ncol] = acc[tt][ct][j];
                }
        }
    }
    __syncthreads();

    // ---- 4) conv along d (zero-pad), 16B-vector bf16 stores ----
    {
        int d8 = (tid & 7) * 8;
        for (int r = tid >> 3; r < RNK; r += 32) {
            float a0 = Ukd[r], a1 = Ukd[256 + r], a2 = Ukd[512 + r];
            v8bf pack;
#pragma unroll
            for (int i = 0; i < 8; ++i) {
                int d = d8 + i;
                float lo = (d > 0)     ? Z[r * ZSTR + d - 1] : 0.f;
                float mi =               Z[r * ZSTR + d];
                float hi = (d < S - 1) ? Z[r * ZSTR + d + 1] : 0.f;
                pack[i] = f2bf(a0 * lo + a1 * mi + a2 * hi);
            }
            *(v8bf*)(t1 + ((size_t)hw * RNK + r) * S + d8) = pack;
        }
    }
}

// ---------------------------------------------------------------------------
// Kernel 2: one workgroup per (h,w) line.
//   1) z[r,d] = sum_{kh,kw} Ukh[kh,r] Ukw[kw,r] t1[h+kh-1,w+kw-1][r][d]
//      9 L2-resident 16B-vector line reads per 8 outputs; staged bf16 in LDS
//   2) out[co,d] = sum_r Ucout[r,co] z[r,d] + bias[co]  via WMMA
//   3) fp32 output staged in LDS, stored as coalesced float4 rows
// LDS: smem time-multiplexed: Bt (bf16 z, 34.8KB) -> Ob (f32 out, 69.6KB).
// ---------------------------------------------------------------------------
__global__ __launch_bounds__(256) void lrconv_k2(const __bf16* __restrict__ t1,
                                                 const float* __restrict__ Ukh,
                                                 const float* __restrict__ Ukw,
                                                 const __bf16* __restrict__ Afrag2,
                                                 const float* __restrict__ bias,
                                                 float* __restrict__ out) {
    __shared__ __align__(32) unsigned char smem[RNK * ZSTR * 4]; // 69632B
    __bf16* Bt = (__bf16*)smem;                                  // [d][r] bf16
    float*  Ob = (float*)smem;                                   // [co][ZSTR] f32

    const int hw = blockIdx.x;
    const int h  = hw >> 6;
    const int w  = hw & 63;
    const int base = hw * S;
    const int tid  = threadIdx.x;
    const int lane = tid & 31;
    const int wave = tid >> 5;
    const int half = lane >> 4;
    const int ncol = lane & 15;

    // ---- 1) h/w conv gather, 16B vector loads (L2-resident t1) ----
    {
        int d8 = (tid & 7) * 8;
        for (int r = tid >> 3; r < RNK; r += 32) {
            float whv[3] = {Ukh[r], Ukh[256 + r], Ukh[512 + r]};
            float wwv[3] = {Ukw[r], Ukw[256 + r], Ukw[512 + r]};
            float acc[8] = {0.f, 0.f, 0.f, 0.f, 0.f, 0.f, 0.f, 0.f};
#pragma unroll
            for (int kh = 0; kh < 3; ++kh) {
                int hh = h + kh - 1;
                if (hh < 0 || hh >= S) continue;
#pragma unroll
                for (int kw = 0; kw < 3; ++kw) {
                    int wc = w + kw - 1;
                    if (wc < 0 || wc >= S) continue;
                    float wgt = whv[kh] * wwv[kw];
                    const v8bf v = *(const v8bf*)(t1 + ((size_t)(hh * S + wc) * RNK + r) * S + d8);
#pragma unroll
                    for (int i = 0; i < 8; ++i)
                        acc[i] += wgt * (float)v[i];
                }
            }
#pragma unroll
            for (int i = 0; i < 8; ++i)
                Bt[(d8 + i) * BSTR + r] = f2bf(acc[i]);
        }
    }
    __syncthreads();

    // ---- 2) GEMM2 ----
    v8f acc[2][4];
#pragma unroll
    for (int tt = 0; tt < 2; ++tt)
#pragma unroll
        for (int ct = 0; ct < 4; ++ct)
            acc[tt][ct] = v8f{0.f, 0.f, 0.f, 0.f, 0.f, 0.f, 0.f, 0.f};

#pragma unroll
    for (int kc = 0; kc < 8; ++kc) {
        v16bf b[4];
#pragma unroll
        for (int ct = 0; ct < 4; ++ct)
            b[ct] = *(const v16bf*)(&Bt[(ct * 16 + ncol) * BSTR + kc * 32 + half * 16]);
#pragma unroll
        for (int tt = 0; tt < 2; ++tt) {
            const int t = wave * 2 + tt;
            const v16bf a = *(const v16bf*)(Afrag2 + ((size_t)(t * 8 + kc) * 32 + lane) * 16);
#pragma unroll
            for (int ct = 0; ct < 4; ++ct)
                acc[tt][ct] = __builtin_amdgcn_wmma_f32_16x16x32_bf16(
                    false, a, false, b[ct], (short)0, acc[tt][ct], false, false);
        }
    }
    __syncthreads();                                      // Bt reads done; reuse as Ob

    // ---- 3) bias add, stage f32 in LDS, coalesced float4 stores ----
#pragma unroll
    for (int tt = 0; tt < 2; ++tt) {
        const int t = wave * 2 + tt;
#pragma unroll
        for (int ct = 0; ct < 4; ++ct)
#pragma unroll
            for (int j = 0; j < 8; ++j) {
                int co = t * 16 + j + 8 * half;
                Ob[co * ZSTR + ct * 16 + ncol] = acc[tt][ct][j] + bias[co];
            }
    }
    __syncthreads();
#pragma unroll
    for (int it = 0; it < 16; ++it) {
        int e  = tid + it * 256;
        int co = e >> 4;
        int d4 = (e & 15) * 4;
        v4f v = {Ob[co * ZSTR + d4],     Ob[co * ZSTR + d4 + 1],
                 Ob[co * ZSTR + d4 + 2], Ob[co * ZSTR + d4 + 3]};
        *(v4f*)(out + (size_t)co * SP + base + d4) = v;
    }
}

// ---------------------------------------------------------------------------
extern "C" void kernel_launch(void* const* d_in, const int* in_sizes, int n_in,
                              void* d_out, int out_size, void* d_ws, size_t ws_size,
                              hipStream_t stream) {
    (void)in_sizes; (void)n_in; (void)out_size; (void)ws_size;
    const float* x     = (const float*)d_in[0];
    const float* Ukh   = (const float*)d_in[1];
    const float* Ukw   = (const float*)d_in[2];
    const float* Ukd   = (const float*)d_in[3];
    const float* Ucin  = (const float*)d_in[4];
    const float* Ucout = (const float*)d_in[5];
    const float* bias  = (const float*)d_in[6];
    float* out = (float*)d_out;

    // workspace: A1 frags (128KB) | A2 frags (128KB) | t1 bf16 (128MB)
    __bf16* A1 = (__bf16*)d_ws;
    __bf16* A2 = A1 + 16 * 8 * 32 * 16;
    __bf16* t1 = A2 + 16 * 8 * 32 * 16;

    lrconv_prep<<<32, 256, 0, stream>>>(Ucin, Ucout, A1, A2);
    lrconv_k1<<<HW, 256, 0, stream>>>(x, Ukd, A1, t1);
    lrconv_k2<<<HW, 256, 0, stream>>>(t1, Ukh, Ukw, A2, bias, out);
}
